// MixedLayerWithArc_74560632259347
// MI455X (gfx1250) — matrix operand order repaired and checked by reading, not compiled, not run on hardware
//
#include <hip/hip_runtime.h>

// ---------------------------------------------------------------------------
// MixedLayerWithArc: per-sample branch-selected 3x3 SAME conv, NCHW fp32.
// bf16 implicit GEMM on v_wmma_f32_16x16x32_bf16; double-buffered LDS tiles
// filled with global_load_async_to_lds_b128 (ASYNCcnt pipelined against the
// wmma stream); f32 accumulate + bias epilogue.
//   ws layout: [Wp bf16: 4*9*256*256]  [Xp bf16 NHWC: 64*32*32*256]
//   needs ws_size >= 38,273,024 bytes.
// ---------------------------------------------------------------------------

typedef __attribute__((ext_vector_type(16))) __bf16 v16bf;
typedef __attribute__((ext_vector_type(8)))  __bf16 v8bf;
typedef __attribute__((ext_vector_type(8)))  float  v8f;

#define NBATCH 64
#define CI     256
#define CO     256
#define NPIX   1024
#define NBR    4

#define WP_ELEMS ((size_t)NBR * 9 * CO * CI)         // 2,359,296
#define XP_ELEMS ((size_t)NBATCH * NPIX * CI)        // 16,777,216

#define LX_ELEMS (10 * 36 * 32)                      // 11,520 bf16 = 23,040 B
#define LW_ELEMS (9 * 64 * 32)                       // 18,432 bf16 = 36,864 B

#if defined(__HIP_DEVICE_COMPILE__) && \
    __has_builtin(__builtin_amdgcn_global_load_async_to_lds_b128) && \
    __has_builtin(__builtin_amdgcn_s_wait_asynccnt)
#define USE_ASYNC_LDS 1
typedef int v4i __attribute__((vector_size(16)));
typedef __attribute__((address_space(1))) v4i gv4i;   // global (AS1) 16B vector
typedef __attribute__((address_space(3))) v4i lv4i;   // LDS    (AS3) 16B vector
#else
#define USE_ASYNC_LDS 0
#endif

// ---- pre-pass: W [a][co][ci][kh][kw] f32 -> Wp [a][tap][co][ci] bf16 -------
__global__ __launch_bounds__(256) void repack_w_kernel(
    const float* __restrict__ W, unsigned short* __restrict__ Wp_) {
  __bf16* Wp = (__bf16*)Wp_;
  unsigned idx = blockIdx.x * 256u + threadIdx.x;   // dest-linear, ci fastest
  unsigned ci = idx & 255u;
  unsigned co = (idx >> 8) & 255u;
  unsigned at = idx >> 16;                          // a*9 + t, 0..35
  unsigned a  = at / 9u;
  unsigned t  = at % 9u;
  Wp[idx] = (__bf16)W[(((a * CO + co) * CI + ci) * 9u) + t];
}

// ---- pre-pass: x NCHW f32 -> Xp NHWC bf16 ---------------------------------
__global__ __launch_bounds__(256) void repack_x_kernel(
    const float* __restrict__ x, unsigned short* __restrict__ Xp_) {
  __bf16* Xp = (__bf16*)Xp_;
  unsigned idx = blockIdx.x * 256u + threadIdx.x;   // src-linear (coalesced f32)
  unsigned p  = idx & 1023u;
  unsigned bc = idx >> 10;
  unsigned ci = bc & 255u;
  unsigned b  = bc >> 8;
  Xp[((size_t)(b * NPIX + p)) * CI + ci] = (__bf16)x[idx];
}

// ---- helper: copy 64B global->LDS ------------------------------------------
__device__ __forceinline__ void stage64(const __bf16* src, __bf16* dst) {
#if USE_ASYNC_LDS
  gv4i* g = (gv4i*)src;
  lv4i* l = (lv4i*)dst;
  __builtin_amdgcn_global_load_async_to_lds_b128(g, l, 0,  0);
  __builtin_amdgcn_global_load_async_to_lds_b128(g, l, 16, 0);
  __builtin_amdgcn_global_load_async_to_lds_b128(g, l, 32, 0);
  __builtin_amdgcn_global_load_async_to_lds_b128(g, l, 48, 0);
#else
  const uint4* s = (const uint4*)src;
  uint4* d = (uint4*)dst;
  uint4 t0 = s[0], t1 = s[1], t2 = s[2], t3 = s[3];
  d[0] = t0; d[1] = t1; d[2] = t2; d[3] = t3;
#endif
}

// ---- helper: issue async stage of one ci-chunk (X tile + W tile) -----------
__device__ __forceinline__ void stage_tiles(const __bf16* __restrict__ xbase,
                                            const __bf16* __restrict__ wbase,
                                            __bf16* __restrict__ lxb,
                                            __bf16* __restrict__ lwb,
                                            int r0, int tid, int ci0) {
  // X tile: 10 halo rows x 34 cols x 32 ci (halo cells pre-zeroed)
  for (int task = tid; task < 340; task += 256) {
    const int lr = task / 34, lc = task % 34;
    const int h = r0 - 1 + lr, w = lc - 1;
    if (((unsigned)h < 32u) && ((unsigned)w < 32u))
      stage64(xbase + ((h * 32 + w) * 256 + ci0), &lxb[(lr * 36 + lc) * 32]);
  }
  // W tile: [tap][co_local 64][ci 32]
  for (int task = tid; task < 576; task += 256) {
    const int t = task >> 6, i = task & 63;
    stage64(wbase + (size_t)(t * CO + i) * CI + ci0, &lwb[task * 32]);
  }
}

// ---- main kernel -----------------------------------------------------------
// block = 256 threads = 8 wave32s; block tile: 64 co x 256 px (8 image rows)
// wave (wm,wn): wm in 0..1 -> 32-row co sub-tile; wn in 0..3 -> 64-px quarter
// wave computes 2(M) x 4(N) tiles of 16x16 -> 8 v8f accumulators
// Software pipeline: async-stage chunk k+1 into buffer ^1 while chunk k's 72
// wmmas run from the current buffer; s_wait_asynccnt 0 + one barrier per iter.
__global__ __launch_bounds__(256) void conv_wmma_kernel(
    const unsigned short* __restrict__ Xp_, const unsigned short* __restrict__ Wp_,
    const int* __restrict__ arc, const float* __restrict__ bias,
    float* __restrict__ out) {
  const __bf16* Xp = (const __bf16*)Xp_;
  const __bf16* Wp = (const __bf16*)Wp_;

  __shared__ __align__(16) __bf16 lx[2][LX_ELEMS];   // 46,080 B
  __shared__ __align__(16) __bf16 lw[2][LW_ELEMS];   // 73,728 B

  const int bz  = blockIdx.z;           // batch
  const int co0 = blockIdx.y * 64;      // co tile base
  const int p0  = blockIdx.x * 256;     // pixel tile base
  const int r0  = blockIdx.x * 8;       // image row base

  const int tid  = threadIdx.x;
  const int lane = tid & 31;
  const int wave = tid >> 5;
  const int wm   = wave >> 2;           // 0..1
  const int wn   = wave & 3;            // 0..3

  const int a = arc[bz];                // selected branch (uniform)

  const int m    = lane & 15;
  const bool hiL = (lane >> 4) != 0;
  const int hiA  = hiL ? 8  : 0;        // A-frag K base per ISA 16-bit layout
  const int hiB  = hiL ? 8  : 0;        // B-frag: same K base as A-frag

  const __bf16* xbase = Xp + (size_t)bz * NPIX * CI;
  const __bf16* wbase = Wp + ((size_t)(a * 9) * CO + co0) * CI;

  // ---- one-time: zero OOB halo cells in both buffers (ci-independent) ----
  for (int task = tid; task < 340; task += 256) {
    const int lr = task / 34, lc = task % 34;
    const int h = r0 - 1 + lr, w = lc - 1;
    if (!(((unsigned)h < 32u) && ((unsigned)w < 32u))) {
      uint4 z = {0u, 0u, 0u, 0u};
      uint4* d0 = (uint4*)&lx[0][(lr * 36 + lc) * 32];
      uint4* d1 = (uint4*)&lx[1][(lr * 36 + lc) * 32];
      d0[0] = z; d0[1] = z; d0[2] = z; d0[3] = z;
      d1[0] = z; d1[1] = z; d1[2] = z; d1[3] = z;
    }
  }

  // ---- per-lane LDS fragment offsets (compile-time imm offsets in loop) ----
  const int lwoff = (wm * 32 + m) * 32 + hiA;
  int lxoff[4];
  {
    const int q0 = wn * 64 + m;          // tt tiles: +0,+16,+32,+48 pixels
    lxoff[0] = (((q0      ) >> 5) * 36 + ((q0      ) & 31)) * 32 + hiB;
    lxoff[1] = (((q0 + 16) >> 5) * 36 + ((q0 + 16) & 31)) * 32 + hiB;
    lxoff[2] = (((q0 + 32) >> 5) * 36 + ((q0 + 32) & 31)) * 32 + hiB;
    lxoff[3] = (((q0 + 48) >> 5) * 36 + ((q0 + 48) & 31)) * 32 + hiB;
  }

  v8f acc[8] = {};   // [f*4 + tt], f = M sub-tile (0: rows +0..15, 1: +16..31)

  // ---- prologue: stage chunk 0 into buffer 0 ----
  stage_tiles(xbase, wbase, lx[0], lw[0], r0, tid, 0);
#if USE_ASYNC_LDS
  __builtin_amdgcn_s_wait_asynccnt(0);
#endif
  __syncthreads();

  for (int chunk = 0; chunk < 8; ++chunk) {
    const int cur = chunk & 1;

    // ---- issue async fill of the next chunk into the other buffer ----
    if (chunk < 7)
      stage_tiles(xbase, wbase, lx[cur ^ 1], lw[cur ^ 1], r0, tid,
                  (chunk + 1) * 32);

    // ---- compute chunk `chunk` from buffer `cur` (overlaps async fill) ----
    const __bf16* lwb = &lw[cur][0] + lwoff;
    const __bf16* lxb = &lx[cur][0];
#pragma unroll
    for (int t = 0; t < 9; ++t) {
      const int kh = t / 3, kw = t % 3;
      const int tw = t * 2048;                 // tap stride in lw (elements)
      v8bf a0lo = *(const v8bf*)(lwb + tw);          // row m,   K hiA..+7
      v8bf a0hi = *(const v8bf*)(lwb + tw + 16);     // row m,   K hiA+16..+23
      v8bf a1lo = *(const v8bf*)(lwb + tw + 512);    // row m+16
      v8bf a1hi = *(const v8bf*)(lwb + tw + 528);
      v16bf a0 = __builtin_shufflevector(a0lo, a0hi,
                   0,1,2,3,4,5,6,7,8,9,10,11,12,13,14,15);
      v16bf a1 = __builtin_shufflevector(a1lo, a1hi,
                   0,1,2,3,4,5,6,7,8,9,10,11,12,13,14,15);
      const int sh = (kh * 36 + kw) * 32;      // tap shift in lx (elements)
#pragma unroll
      for (int tt = 0; tt < 4; ++tt) {
        v8bf blo = *(const v8bf*)(lxb + lxoff[tt] + sh);
        v8bf bhi = *(const v8bf*)(lxb + lxoff[tt] + sh + 16);
        v16bf bf = __builtin_shufflevector(blo, bhi,
                     0,1,2,3,4,5,6,7,8,9,10,11,12,13,14,15);
        acc[tt]     = __builtin_amdgcn_wmma_f32_16x16x32_bf16(
            false, a0, false, bf, (short)0, acc[tt],     false, false);
        acc[4 + tt] = __builtin_amdgcn_wmma_f32_16x16x32_bf16(
            false, a1, false, bf, (short)0, acc[4 + tt], false, false);
      }
    }

    // ---- next buffer landed? (in-order ASYNCcnt) + release cur buffer ----
#if USE_ASYNC_LDS
    __builtin_amdgcn_s_wait_asynccnt(0);
#endif
    __syncthreads();
  }

  // ---- epilogue: bias + store (C/D layout: VGPR r -> M=r / M=r+8) ----
  float bv[2][8];
#pragma unroll
  for (int f = 0; f < 2; ++f)
#pragma unroll
    for (int vr = 0; vr < 8; ++vr) {
      const int co = co0 + wm * 32 + f * 16 + vr + (hiL ? 8 : 0);
      bv[f][vr] = bias[a * CO + co];
    }
#pragma unroll
  for (int f = 0; f < 2; ++f) {
#pragma unroll
    for (int tt = 0; tt < 4; ++tt) {
      const int n = p0 + wn * 64 + tt * 16 + m;
#pragma unroll
      for (int vr = 0; vr < 8; ++vr) {
        const int co = co0 + wm * 32 + f * 16 + vr + (hiL ? 8 : 0);
        out[((size_t)(bz * CO + co)) * NPIX + n] = acc[f * 4 + tt][vr] + bv[f][vr];
      }
    }
  }
}

// ---------------------------------------------------------------------------
extern "C" void kernel_launch(void* const* d_in, const int* in_sizes, int n_in,
                              void* d_out, int out_size, void* d_ws, size_t ws_size,
                              hipStream_t stream) {
  (void)in_sizes; (void)n_in; (void)out_size; (void)ws_size;
  const float* x    = (const float*)d_in[0];
  const int*   arc  = (const int*)d_in[1];
  const float* W    = (const float*)d_in[2];
  const float* bias = (const float*)d_in[3];
  float* out = (float*)d_out;

  unsigned short* Wp = (unsigned short*)d_ws;
  unsigned short* Xp = (unsigned short*)((char*)d_ws + WP_ELEMS * sizeof(unsigned short));

  repack_w_kernel<<<dim3((unsigned)(WP_ELEMS / 256)), dim3(256), 0, stream>>>(W, Wp);
  repack_x_kernel<<<dim3((unsigned)(XP_ELEMS / 256)), dim3(256), 0, stream>>>(x, Xp);
  conv_wmma_kernel<<<dim3(4, 4, 64), dim3(256), 0, stream>>>(Xp, Wp, arc, bias, out);
}
